// RGCN_7627861918258
// MI455X (gfx1250) — compile-verified, matrix operand order checked by reading, not run on hardware
//
#include <hip/hip_runtime.h>
#include <hip/hip_bf16.h>

typedef __attribute__((ext_vector_type(16))) _Float16 v16h;
typedef __attribute__((ext_vector_type(8)))  _Float16 v8h;
typedef __attribute__((ext_vector_type(8)))  float    v8f;

#define Nn   50000
#define Rr   51
#define Ff   32
#define Ee   32
#define Cc   16
#define Bb   30
#define NNZk 2000000

// ---------------------------------------------------------------------------
// K0: basis combine w[r][k][n] = sum_b comps[r,b]*bases[b,k,n], packed directly
// into the CDNA5 WMMA 16-bit B-matrix lane layout so the GEMM kernel can load
// each lane's v16h with two contiguous b128 loads.
// Layout: wpk[((r*nebs+eb)*32 + lane)*16 + j]  holds  w[r][kmap(lane,j)][eb*16+(lane&15)]
// kmap(lane,j) = (j<8 ? j : j+8) + ((lane>>4)<<3)     (ISA 7.12.2, 16-bit A/B)
// ---------------------------------------------------------------------------
__global__ void pack_w_kernel(const float* __restrict__ comps,
                              const float* __restrict__ bases,
                              _Float16* __restrict__ wpk,
                              int Kdim, int Ncols, int nebs) {
    int r = blockIdx.x;
    int total = nebs * 32 * 16;
    for (int t = threadIdx.x; t < total; t += blockDim.x) {
        int j    = t & 15;
        int lane = (t >> 4) & 31;
        int eb   = t >> 9;
        int n    = eb * 16 + (lane & 15);
        int k    = (j < 8 ? j : j + 8) + ((lane >> 4) << 3);
        float acc = 0.f;
        #pragma unroll
        for (int b = 0; b < Bb; ++b)
            acc += comps[r * Bb + b] * bases[((size_t)b * Kdim + k) * Ncols + n];
        wpk[((size_t)(r * nebs + eb) * 32 + lane) * 16 + j] = (_Float16)acc;
    }
}

__global__ void f32_to_f16_kernel(const float* __restrict__ in,
                                  _Float16* __restrict__ out, int n) {
    int i = blockIdx.x * blockDim.x + threadIdx.x;
    if (i < n) out[i] = (_Float16)in[i];
}

__global__ void init_zero_kernel(float* __restrict__ p, int n) {
    int i = blockIdx.x * blockDim.x + threadIdx.x;
    if (i < n) p[i] = 0.f;
}

__global__ void init_bias_kernel(float* __restrict__ out,
                                 const float* __restrict__ bias, int n) {
    int i = blockIdx.x * blockDim.x + threadIdx.x;
    if (i < n) out[i] = bias[i & 15];           // cols = 16
}

__global__ void relu_bias_cvt_kernel(const float* __restrict__ h,
                                     const float* __restrict__ bias,
                                     _Float16* __restrict__ out, int n) {
    int i = blockIdx.x * blockDim.x + threadIdx.x;
    if (i < n) {
        float x = h[i] + bias[i & 31];          // cols = 32
        out[i] = (_Float16)(x > 0.f ? x : 0.f);
    }
}

// ---------------------------------------------------------------------------
// Dense precompute z[r][m][:] = a16[m,:] @ W_r via v_wmma_f32_16x16x32_f16.
// One wave owns a 16-node tile; A (16x32 f16) is loaded once into registers
// (ISA A-layout: lane<16 holds K 0-7/16-23, lane>=16 holds K 8-15/24-31 of
// row M=lane&15), then it loops over all R*nebs pre-packed B tiles: exactly
// one WMMA per output 16x16 tile since K == 32.
// ---------------------------------------------------------------------------
template <int NEBS>
__global__ void wmma_gemm_kernel(const _Float16* __restrict__ a16,   // (Nn x 32)
                                 const _Float16* __restrict__ wpk,   // packed B
                                 _Float16* __restrict__ zout) {      // (R, Nn, NEBS*16)
    const int nNodeTiles = Nn / 16;                                  // 3125
    int wave = blockIdx.x * (blockDim.x >> 5) + (threadIdx.x >> 5);
    if (wave >= nNodeTiles) return;
    const int outCols = NEBS * 16;
    int lane = threadIdx.x & 31;
    int n0   = wave * 16;
    int kc   = lane >> 4;                      // which K half this lane holds

    const v8h* rowp = (const v8h*)(a16 + (size_t)(n0 + (lane & 15)) * 32);
    union { v16h v; v8h h[2]; } A;
    A.h[0] = rowp[kc];                         // K  0-7  or  8-15
    A.h[1] = rowp[2 + kc];                     // K 16-23 or 24-31

    for (int tb = 0; tb < Rr * NEBS; ++tb) {
        const v8h* bp = (const v8h*)(wpk + ((size_t)tb * 32 + lane) * 16);
        union { v16h v; v8h h[2]; } Bm;
        Bm.h[0] = bp[0];
        Bm.h[1] = bp[1];
        v8f c = {};
        c = __builtin_amdgcn_wmma_f32_16x16x32_f16(
                /*neg_a=*/false, A.v, /*neg_b=*/false, Bm.v,
                /*c_mod=*/(short)0, c, /*reuse_a=*/false, /*reuse_b=*/false);
        int r  = tb / NEBS;
        int eb = tb - r * NEBS;
        int e  = eb * 16 + (lane & 15);        // D col = lane&15
        // D layout: lane<16 -> rows v+0..7 ; lane>=16 -> rows v+8..15
        size_t base = ((size_t)r * Nn + n0 + kc * 8) * outCols + e;
        #pragma unroll
        for (int v = 0; v < 8; ++v)
            zout[base + (size_t)v * outCols] = (_Float16)c[v];
    }
}

// ---------------------------------------------------------------------------
// Sparse scatter, layer 1: one wave per edge, lane = output feature (32).
// out1[n,lane] += val * z1[r][m][lane];  rows[i] encodes r*N + n.
// ---------------------------------------------------------------------------
__global__ void edge_acc32_kernel(const int* __restrict__ rows,
                                  const int* __restrict__ cols,
                                  const float* __restrict__ vals,
                                  const _Float16* __restrict__ z,
                                  float* __restrict__ acc) {
    int lane   = threadIdx.x & 31;
    int wave   = blockIdx.x * (blockDim.x >> 5) + (threadIdx.x >> 5);
    int nwaves = gridDim.x * (blockDim.x >> 5);
    for (int i = wave; i < NNZk; i += nwaves) {
        __builtin_prefetch(rows + i + nwaves, 0, 1);
        __builtin_prefetch(cols + i + nwaves, 0, 1);
        int   rr = rows[i];
        int   m  = cols[i];
        float v  = vals[i];
        int   r  = rr / Nn;
        int   n  = rr - r * Nn;
        float zv = (float)z[((size_t)r * Nn + m) * 32 + lane];
        atomicAdd(acc + (size_t)n * 32 + lane, v * zv);
    }
}

// Sparse scatter, layer 2: half-wave per edge, lane&15 = output class (16).
__global__ void edge_acc16_kernel(const int* __restrict__ rows,
                                  const int* __restrict__ cols,
                                  const float* __restrict__ vals,
                                  const _Float16* __restrict__ z,
                                  float* __restrict__ out) {
    int lane   = threadIdx.x & 31;
    int half   = lane >> 4;
    int c      = lane & 15;
    int wave   = blockIdx.x * (blockDim.x >> 5) + (threadIdx.x >> 5);
    int nwaves = gridDim.x * (blockDim.x >> 5);
    for (int i = wave * 2 + half; i < NNZk; i += nwaves * 2) {
        int   rr = rows[i];
        int   m  = cols[i];
        float v  = vals[i];
        int   r  = rr / Nn;
        int   n  = rr - r * Nn;
        float zv = (float)z[((size_t)r * Nn + m) * 16 + c];
        atomicAdd(out + (size_t)n * 16 + c, v * zv);
    }
}

extern "C" void kernel_launch(void* const* d_in, const int* in_sizes, int n_in,
                              void* d_out, int out_size, void* d_ws, size_t ws_size,
                              hipStream_t stream) {
    const float* features = (const float*)d_in[0];
    const float* vals     = (const float*)d_in[1];
    const float* comps1   = (const float*)d_in[2];
    const float* bases1   = (const float*)d_in[3];
    const float* bias1    = (const float*)d_in[4];
    const float* comps2   = (const float*)d_in[5];
    const float* bases2   = (const float*)d_in[6];
    const float* bias2    = (const float*)d_in[7];
    const int*   rows     = (const int*)d_in[8];
    const int*   cols     = (const int*)d_in[9];
    float*       out      = (float*)d_out;

    char*  ws  = (char*)d_ws;
    size_t off = 0;
    auto alloc = [&](size_t bytes) {
        void* p = ws + off;
        off = (off + bytes + 255) & ~(size_t)255;
        return p;
    };
    _Float16* w1pk   = (_Float16*)alloc((size_t)Rr * 2 * 32 * 16 * 2);
    _Float16* w2pk   = (_Float16*)alloc((size_t)Rr * 1 * 32 * 16 * 2);
    _Float16* feat16 = (_Float16*)alloc((size_t)Nn * 32 * 2);
    _Float16* h1_16  = (_Float16*)alloc((size_t)Nn * 32 * 2);
    float*    h1     = (float*)alloc((size_t)Nn * 32 * 4);
    _Float16* zbuf   = (_Float16*)alloc((size_t)Rr * Nn * 32 * 2);  // reused: z1 then z2

    const int NF = Nn * 32;   // 1.6M
    const int NC = Nn * 16;

    // weight combine + pack into WMMA B layout
    pack_w_kernel<<<Rr, 256, 0, stream>>>(comps1, bases1, w1pk, 32, 32, 2);
    pack_w_kernel<<<Rr, 256, 0, stream>>>(comps2, bases2, w2pk, 32, 16, 1);
    // feature f16 conversion + accumulator init
    f32_to_f16_kernel<<<(NF + 255) / 256, 256, 0, stream>>>(features, feat16, NF);
    init_zero_kernel <<<(NF + 255) / 256, 256, 0, stream>>>(h1, NF);
    // layer 1: dense WMMA precompute z1 = feat @ W_r  -> edge scatter
    wmma_gemm_kernel<2><<<(3125 + 7) / 8, 256, 0, stream>>>(feat16, w1pk, zbuf);
    edge_acc32_kernel<<<2048, 256, 0, stream>>>(rows, cols, vals, zbuf, h1);
    // activation
    relu_bias_cvt_kernel<<<(NF + 255) / 256, 256, 0, stream>>>(h1, bias1, h1_16, NF);
    // layer 2: dense WMMA precompute z2 = h1 @ W2_r -> edge scatter (+bias init)
    wmma_gemm_kernel<1><<<(3125 + 7) / 8, 256, 0, stream>>>(h1_16, w2pk, zbuf);
    init_bias_kernel<<<(NC + 255) / 256, 256, 0, stream>>>(out, bias2, NC);
    edge_acc16_kernel<<<2048, 256, 0, stream>>>(rows, cols, vals, zbuf, out);
}
